// FrontierLayerVN_16338055594664
// MI455X (gfx1250) — matrix operand order repaired and checked by reading, not compiled, not run on hardware
//
#include <hip/hip_runtime.h>
#include <hip/hip_bf16.h>
#include <math.h>

typedef __attribute__((ext_vector_type(16))) _Float16 v16h;
typedef __attribute__((ext_vector_type(8)))  _Float16 v8h;
typedef __attribute__((ext_vector_type(8)))  float    v8f;

#define WPB 4   // waves per block (wave32)

// ---- fragment table offsets (in halves) inside d_ws ------------------------
// Each B-fragment tile = 32 lanes x 16 halves = 512 halves.
// Lane layout (dense f16 B, KxN=32x16): N = lane%16, half j -> K = 8*(lane/16)+(j&7)+16*(j>>3).
#define F_WV11 0        // Wv1_1: K=64,N=64   -> 2 kc x 4 nt = 8 tiles
#define F_WS1  4096     // Ws_1 : K=320,N=128 -> 10 kc x 8 nt = 80 tiles
#define F_WG1  45056    // Wg_1 : K=128,N=32  -> 4 kc x 2 nt = 8 tiles
#define F_WV21 49152    // Wv2_1: K=64,N=32   -> 2 kc x 2 nt = 4 tiles
#define F_WD1  51200    // Wd_1 : K=32,N=32   -> 2 tiles
#define F_WV12 52224    // Wv1_2: K=32,N=32   -> 2 tiles
#define FRAG_TOTAL 53248

__global__ void prep_frags(const float* __restrict__ Wv11, const float* __restrict__ Ws1,
                           const float* __restrict__ Wg1,  const float* __restrict__ Wv21,
                           const float* __restrict__ Wd1,  const float* __restrict__ Wv12,
                           _Float16* __restrict__ frags) {
  int g = blockIdx.x * blockDim.x + threadIdx.x;
  if (g >= FRAG_TOTAL) return;
  const float* src; int NT, srcN, local;
  if      (g < F_WS1)  { src = Wv11; NT = 4; srcN = 64;  local = g; }
  else if (g < F_WG1)  { src = Ws1;  NT = 8; srcN = 128; local = g - F_WS1; }
  else if (g < F_WV21) { src = Wg1;  NT = 2; srcN = 32;  local = g - F_WG1; }
  else if (g < F_WD1)  { src = Wv21; NT = 2; srcN = 32;  local = g - F_WV21; }
  else if (g < F_WV12) { src = Wd1;  NT = 2; srcN = 32;  local = g - F_WD1; }
  else                 { src = Wv12; NT = 2; srcN = 32;  local = g - F_WV12; }
  int tile = local >> 9;
  int r    = local & 511;
  int ln   = r >> 4;       // lane 0..31
  int j    = r & 15;       // half index within lane
  int kc = tile / NT, nt = tile % NT;
  int K = kc * 32 + ((ln >> 4) << 3) + (j & 7) + ((j >> 3) << 4);
  int N = nt * 16 + (ln & 15);
  frags[g] = (_Float16)src[K * srcN + N];
}

// ---- device helpers --------------------------------------------------------
__device__ __forceinline__ v8f wmma_step(v16h a, v16h b, v8f c) {
  return __builtin_amdgcn_wmma_f32_16x16x32_f16(false, a, false, b, (short)0, c, false, false);
}

__device__ __forceinline__ v16h b_frag(const _Float16* __restrict__ frags,
                                       int half_off, int tile, int lane) {
  return *(const v16h*)(frags + half_off + (((size_t)(tile * 32 + lane)) << 4));
}

// A-fragment (16x32 f16): lane L -> row M=L&15; halves 0-7 = K base0..base0+7,
// halves 8-15 = K base0+16..base0+23, base0 = (L>=16 ? 8 : 0).
__device__ __forceinline__ v16h a_from_lds_h(const _Float16* __restrict__ base,
                                             int ld, int kbase, int lane) {
  int m = lane & 15, hb = (lane >> 4) << 3;
  const _Float16* p = base + m * ld + kbase + hb;
  v8h x = *(const v8h*)p;
  v8h y = *(const v8h*)(p + 16);
  v16h a;
#pragma unroll
  for (int t = 0; t < 8; t++) { a[t] = x[t]; a[t + 8] = y[t]; }
  return a;
}

// ---- main kernel: one wave per 16-row tile ---------------------------------
__global__ __launch_bounds__(WPB * 32)
void frontier_kernel(const float* __restrict__ sca_in, const float* __restrict__ vec_in,
                     const int* __restrict__ idx, const float* __restrict__ bg1,
                     const float* __restrict__ Ws2, const _Float16* __restrict__ frags,
                     float* __restrict__ out, int M) {
  __shared__ __align__(16) _Float16 s_vh[WPB][3][16 * 64];  // vh scratch (also reused ld=32)
  __shared__ __align__(16) _Float16 s_vn[WPB][16 * 64];     // vnorm scratch
  __shared__ __align__(16) _Float16 s_sca[WPB][16 * 128];   // out_sca1 scratch
  __shared__ __align__(16) float    s_part[WPB][16 * 16];   // stage-I partials

  const int lane  = threadIdx.x & 31;
  const int wid   = threadIdx.x >> 5;
  const int mloc  = lane & 15;
  const int mbase = (lane < 16) ? 0 : 8;
  const int hb    = (lane >> 4) << 3;

  const int ntiles = (M + 15) >> 4;
  for (int tile = blockIdx.x * WPB + wid; tile < ntiles; tile += gridDim.x * WPB) {
    int mrow = tile * 16 + mloc;
    int gidx = idx[(mrow < M) ? mrow : (M - 1)];
    const float* srow = sca_in + (size_t)gidx * 256;

    // ---- stage A: vh[16x64x3] = vec @ Wv1_1 (WMMA, 24x) ----
    v8f vh[3][4];
#pragma unroll
    for (int i = 0; i < 3; i++)
#pragma unroll
      for (int nt = 0; nt < 4; nt++) vh[i][nt] = v8f{};
#pragma unroll
    for (int i = 0; i < 3; i++) {
      const float* vrow = vec_in + (size_t)gidx * 192 + i;  // element c at vrow[c*3]
#pragma unroll
      for (int kc = 0; kc < 2; kc++) {
        v16h a;
#pragma unroll
        for (int t = 0; t < 8; t++) {
          a[t]     = (_Float16)vrow[(kc * 32 + hb + t) * 3];
          a[t + 8] = (_Float16)vrow[(kc * 32 + hb + 16 + t) * 3];
        }
#pragma unroll
        for (int nt = 0; nt < 4; nt++)
          vh[i][nt] = wmma_step(a, b_frag(frags, F_WV11, kc * 4 + nt, lane), vh[i][nt]);
      }
    }

    // ---- stage B: vnorm + spill vh/vnorm to LDS (f16) ----
#pragma unroll
    for (int nt = 0; nt < 4; nt++) {
#pragma unroll
      for (int e = 0; e < 8; e++) {
        float x = vh[0][nt][e], y = vh[1][nt][e], z = vh[2][nt][e];
        int off = (e + mbase) * 64 + nt * 16 + mloc;
        s_vn[wid][off]    = (_Float16)sqrtf(x * x + y * y + z * z);
        s_vh[wid][0][off] = (_Float16)x;
        s_vh[wid][1][off] = (_Float16)y;
        s_vh[wid][2][off] = (_Float16)z;
      }
    }
    asm volatile("s_wait_dscnt 0" ::: "memory");

    // ---- stage C: out_sca1[16x128] = [vnorm|sca](320) @ Ws_1 (WMMA, 80x) ----
    v8f sc[8];
#pragma unroll
    for (int nt = 0; nt < 8; nt++) sc[nt] = v8f{};
#pragma unroll
    for (int kc = 0; kc < 2; kc++) {
      v16h a = a_from_lds_h(&s_vn[wid][0], 64, kc * 32, lane);
#pragma unroll
      for (int nt = 0; nt < 8; nt++)
        sc[nt] = wmma_step(a, b_frag(frags, F_WS1, kc * 8 + nt, lane), sc[nt]);
    }
#pragma unroll
    for (int kc = 0; kc < 8; kc++) {
      const float* p = srow + kc * 32 + hb;
      float4 f0 = *(const float4*)(p);
      float4 f1 = *(const float4*)(p + 4);
      float4 f2 = *(const float4*)(p + 16);
      float4 f3 = *(const float4*)(p + 20);
      v16h a;
      a[0]=(_Float16)f0.x; a[1]=(_Float16)f0.y; a[2]=(_Float16)f0.z; a[3]=(_Float16)f0.w;
      a[4]=(_Float16)f1.x; a[5]=(_Float16)f1.y; a[6]=(_Float16)f1.z; a[7]=(_Float16)f1.w;
      a[8]=(_Float16)f2.x; a[9]=(_Float16)f2.y; a[10]=(_Float16)f2.z; a[11]=(_Float16)f2.w;
      a[12]=(_Float16)f3.x; a[13]=(_Float16)f3.y; a[14]=(_Float16)f3.z; a[15]=(_Float16)f3.w;
#pragma unroll
      for (int nt = 0; nt < 8; nt++)
        sc[nt] = wmma_step(a, b_frag(frags, F_WS1, (kc + 2) * 8 + nt, lane), sc[nt]);
    }
#pragma unroll
    for (int nt = 0; nt < 8; nt++)
#pragma unroll
      for (int e = 0; e < 8; e++)
        s_sca[wid][(e + mbase) * 128 + nt * 16 + mloc] = (_Float16)sc[nt][e];
    asm volatile("s_wait_dscnt 0" ::: "memory");

    // ---- stage E: gate = sigmoid(out_sca1 @ Wg_1 + bg_1) (WMMA, 8x) ----
    v8f g2[2]; g2[0] = v8f{}; g2[1] = v8f{};
#pragma unroll
    for (int kc = 0; kc < 4; kc++) {
      v16h a = a_from_lds_h(&s_sca[wid][0], 128, kc * 32, lane);
#pragma unroll
      for (int nt = 0; nt < 2; nt++)
        g2[nt] = wmma_step(a, b_frag(frags, F_WG1, kc * 2 + nt, lane), g2[nt]);
    }
    float gate[2][8];
#pragma unroll
    for (int nt = 0; nt < 2; nt++) {
      float b = bg1[nt * 16 + mloc];
#pragma unroll
      for (int e = 0; e < 8; e++)
        gate[nt][e] = 1.0f / (1.0f + __expf(-(g2[nt][e] + b)));
    }

    // ---- stage D: out_vec1 = (vh @ Wv2_1) * gate (WMMA, 12x) ----
    v8f v1[3][2];
#pragma unroll
    for (int i = 0; i < 3; i++) { v1[i][0] = v8f{}; v1[i][1] = v8f{}; }
#pragma unroll
    for (int i = 0; i < 3; i++)
#pragma unroll
      for (int kc = 0; kc < 2; kc++) {
        v16h a = a_from_lds_h(&s_vh[wid][i][0], 64, kc * 32, lane);
#pragma unroll
        for (int nt = 0; nt < 2; nt++)
          v1[i][nt] = wmma_step(a, b_frag(frags, F_WV21, kc * 2 + nt, lane), v1[i][nt]);
      }
#pragma unroll
    for (int i = 0; i < 3; i++)
#pragma unroll
      for (int nt = 0; nt < 2; nt++)
#pragma unroll
        for (int e = 0; e < 8; e++) v1[i][nt][e] *= gate[nt][e];

    // ---- stage F: VN-LeakyReLU (d = vec1 @ Wd_1 via WMMA, 6x) ----
#pragma unroll
    for (int i = 0; i < 3; i++)
#pragma unroll
      for (int nt = 0; nt < 2; nt++)
#pragma unroll
        for (int e = 0; e < 8; e++)
          s_vh[wid][i][(e + mbase) * 32 + nt * 16 + mloc] = (_Float16)v1[i][nt][e];
    asm volatile("s_wait_dscnt 0" ::: "memory");
    v8f dm[3][2];
#pragma unroll
    for (int i = 0; i < 3; i++) {
      dm[i][0] = v8f{}; dm[i][1] = v8f{};
      v16h a = a_from_lds_h(&s_vh[wid][i][0], 32, 0, lane);
#pragma unroll
      for (int nt = 0; nt < 2; nt++)
        dm[i][nt] = wmma_step(a, b_frag(frags, F_WD1, nt, lane), dm[i][nt]);
    }
    v8f v2[3][2];
#pragma unroll
    for (int nt = 0; nt < 2; nt++)
#pragma unroll
      for (int e = 0; e < 8; e++) {
        float dot = v1[0][nt][e] * dm[0][nt][e] + v1[1][nt][e] * dm[1][nt][e] +
                    v1[2][nt][e] * dm[2][nt][e];
        float dsq = dm[0][nt][e] * dm[0][nt][e] + dm[1][nt][e] * dm[1][nt][e] +
                    dm[2][nt][e] * dm[2][nt][e];
        float s = dot / (dsq + 1e-6f);
#pragma unroll
        for (int i = 0; i < 3; i++) {
          float v = v1[i][nt][e], d = dm[i][nt][e];
          float keep = (dot >= 0.0f) ? v : (v - s * d);
          v2[i][nt][e] = 0.2f * v + 0.8f * keep;
        }
      }

    // ---- stage H: vh2 = vec2 @ Wv1_2 (WMMA, 6x); vnorm2 ----
#pragma unroll
    for (int i = 0; i < 3; i++)
#pragma unroll
      for (int nt = 0; nt < 2; nt++)
#pragma unroll
        for (int e = 0; e < 8; e++)
          s_vh[wid][i][(e + mbase) * 32 + nt * 16 + mloc] = (_Float16)v2[i][nt][e];
    asm volatile("s_wait_dscnt 0" ::: "memory");
    v8f h2[3][2];
#pragma unroll
    for (int i = 0; i < 3; i++) {
      h2[i][0] = v8f{}; h2[i][1] = v8f{};
      v16h a = a_from_lds_h(&s_vh[wid][i][0], 32, 0, lane);
#pragma unroll
      for (int nt = 0; nt < 2; nt++)
        h2[i][nt] = wmma_step(a, b_frag(frags, F_WV12, nt, lane), h2[i][nt]);
    }

    // ---- stage I: out[m] = vnorm2 . Ws2[0:32] + leaky(sca) . Ws2[32:160] ----
    float w0 = Ws2[mloc], w1 = Ws2[16 + mloc];
#pragma unroll
    for (int e = 0; e < 8; e++) {
      float n0 = sqrtf(h2[0][0][e] * h2[0][0][e] + h2[1][0][e] * h2[1][0][e] +
                       h2[2][0][e] * h2[2][0][e]);
      float n1 = sqrtf(h2[0][1][e] * h2[0][1][e] + h2[1][1][e] * h2[1][1][e] +
                       h2[2][1][e] * h2[2][1][e]);
      s_part[wid][(e + mbase) * 16 + mloc] = n0 * w0 + n1 * w1;
    }
    asm volatile("s_wait_dscnt 0" ::: "memory");
    if (lane < 16) {
      int m = lane;
      float acc = 0.0f;
#pragma unroll
      for (int n = 0; n < 16; n++) acc += s_part[wid][m * 16 + n];
      for (int k = 0; k < 128; k++) {
        float s = (float)s_sca[wid][m * 128 + k];
        s = (s >= 0.0f) ? s : 0.01f * s;
        acc += s * Ws2[32 + k];
      }
      int orow = tile * 16 + m;
      if (orow < M) out[orow] = acc;
    }
  }
}

// ---- host entry ------------------------------------------------------------
extern "C" void kernel_launch(void* const* d_in, const int* in_sizes, int n_in,
                              void* d_out, int out_size, void* d_ws, size_t ws_size,
                              hipStream_t stream) {
  (void)n_in; (void)out_size; (void)ws_size;
  const float* sca   = (const float*)d_in[0];
  const float* vec   = (const float*)d_in[1];
  const int*   idx   = (const int*)d_in[2];
  const float* Wv1_1 = (const float*)d_in[3];
  const float* Wv2_1 = (const float*)d_in[4];
  const float* Ws_1  = (const float*)d_in[5];
  const float* Wg_1  = (const float*)d_in[6];
  const float* bg_1  = (const float*)d_in[7];
  const float* Wd_1  = (const float*)d_in[8];
  const float* Wv1_2 = (const float*)d_in[9];
  const float* Ws_2  = (const float*)d_in[11];
  float*       out   = (float*)d_out;
  _Float16*    frags = (_Float16*)d_ws;

  int M = in_sizes[2];

  prep_frags<<<(FRAG_TOTAL + 255) / 256, 256, 0, stream>>>(
      Wv1_1, Ws_1, Wg_1, Wv2_1, Wd_1, Wv1_2, frags);

  int ntiles = (M + 15) >> 4;
  int blocks = (ntiles + WPB - 1) / WPB;
  frontier_kernel<<<blocks, WPB * 32, 0, stream>>>(
      sca, vec, idx, bg_1, Ws_2, frags, out, M);
}
